// Geo_branch_36395552866673
// MI455X (gfx1250) — matrix operand order repaired and checked by reading, not compiled, not run on hardware
//
#include <hip/hip_runtime.h>
#include <hip/hip_bf16.h>

// ---------------------------------------------------------------------------
// CDNA5 (gfx1250) implementation: wave32 WMMA bf16 GEMMs + atomic segment
// softmax for graph attention.
// ---------------------------------------------------------------------------

typedef __attribute__((ext_vector_type(16))) __bf16 v16bf;
typedef __attribute__((ext_vector_type(8)))  float  v8f;

// ---- A fragment: 16x32 bf16, lane<16 takes K {kt..kt+7, kt+16..kt+23},
// ---- lane>=16 takes K {kt+8..kt+15, kt+24..kt+31}  (ISA 7.12.2 layout)
__device__ __forceinline__ v16bf load_afrag(const float* __restrict__ p, int half) {
  const float4 a = *(const float4*)(p + half * 8);
  const float4 b = *(const float4*)(p + half * 8 + 4);
  const float4 c = *(const float4*)(p + 16 + half * 8);
  const float4 d = *(const float4*)(p + 16 + half * 8 + 4);
  v16bf f;
  f[0]  = (__bf16)a.x; f[1]  = (__bf16)a.y; f[2]  = (__bf16)a.z; f[3]  = (__bf16)a.w;
  f[4]  = (__bf16)b.x; f[5]  = (__bf16)b.y; f[6]  = (__bf16)b.z; f[7]  = (__bf16)b.w;
  f[8]  = (__bf16)c.x; f[9]  = (__bf16)c.y; f[10] = (__bf16)c.z; f[11] = (__bf16)c.w;
  f[12] = (__bf16)d.x; f[13] = (__bf16)d.y; f[14] = (__bf16)d.z; f[15] = (__bf16)d.w;
  return f;
}

// ---- B fragment: 32x16 bf16 from LDS-staged f32 weights [Ktot,32] row-major.
// lane = column; lanes 0-15 hold K rows {krow0..krow0+7, krow0+16..krow0+23}, lanes 16-31 start at +8.
__device__ __forceinline__ v16bf load_bfrag(const float* __restrict__ sW, int krow0, int col) {
  v16bf f;
#pragma unroll
  for (int e = 0; e < 16; ++e) f[e] = (__bf16)sW[(krow0 + e + (e & 8)) * 32 + col];
  return f;
}

__device__ __forceinline__ void mma_tile(const float* __restrict__ r,
                                         const float* __restrict__ sW,
                                         int krow, int half, int lr,
                                         v8f& acc0, v8f& acc1) {
  v16bf A  = load_afrag(r, half);
  v16bf B0 = load_bfrag(sW, krow + half * 8, lr);
  v16bf B1 = load_bfrag(sW, krow + half * 8, lr + 16);
  acc0 = __builtin_amdgcn_wmma_f32_16x16x32_bf16(false, A, false, B0, (short)0, acc0, false, false);
  acc1 = __builtin_amdgcn_wmma_f32_16x16x32_bf16(false, A, false, B1, (short)0, acc1, false, false);
}

// ---------------------------------------------------------------------------
// Flexible WMMA GEMM:  Out[i,:] = concat(X0[g(i),:cin0], X1[i,:32], X2[i,:32]) @ W
// epi: 0 = plain store  1 = LayerNorm+ReLU (g,b)  2 = +bias, 20-col output
// ---------------------------------------------------------------------------
__global__ __launch_bounds__(128)
void wmma_gemm(const float* __restrict__ X0, const int* __restrict__ G0, int cin0,
               const float* __restrict__ X1, int cin1,
               const float* __restrict__ X2, int cin2,
               const float* __restrict__ W, int N, float* __restrict__ Out,
               int epi, const float* __restrict__ gv, const float* __restrict__ bv) {
  __shared__ float sW[224 * 32];
  const int Ktot = cin0 + cin1 + cin2;
  for (int i = threadIdx.x; i < Ktot * 32; i += 128) sW[i] = W[i];
  __syncthreads();

  const int wave = threadIdx.x >> 5;
  const int lane = threadIdx.x & 31;
  const int half = lane >> 4;
  const int lr   = lane & 15;
  const int m0   = (blockIdx.x * 4 + wave) * 16;
  if (m0 >= N) return;  // wave-uniform: EXEC stays all-ones for WMMA

  const int  arow   = m0 + lr;
  const bool avalid = arow < N;
  const int  g0     = avalid ? (G0 ? G0[arow] : arow) : 0;
  const float* r0 = X0 + (size_t)g0 * cin0;
  const float* r1 = X1 ? (X1 + (size_t)(avalid ? arow : 0) * 32) : nullptr;
  const float* r2 = X2 ? (X2 + (size_t)(avalid ? arow : 0) * 32) : nullptr;

  v8f acc0 = {}, acc1 = {};
  int kbase = 0;
  for (int kt = 0; kt < cin0; kt += 32)
    mma_tile(r0 + kt, sW, kbase + kt, half, lr, acc0, acc1);
  kbase += cin0;
  if (r1) { mma_tile(r1, sW, kbase, half, lr, acc0, acc1); kbase += cin1; }
  if (r2) { mma_tile(r2, sW, kbase, half, lr, acc0, acc1); }

  // C/D layout: element v of acc -> row m0 + 8*half + v, col lr (acc0) / 16+lr (acc1)
  if (epi == 1) {
    const float gl0 = gv[lr], gl1 = gv[lr + 16];
    const float bl0 = bv[lr], bl1 = bv[lr + 16];
#pragma unroll
    for (int v = 0; v < 8; ++v) {
      const int row = m0 + half * 8 + v;
      const float x0 = acc0[v], x1 = acc1[v];
      float s = x0 + x1;
#pragma unroll
      for (int o = 1; o < 16; o <<= 1) s += __shfl_xor(s, o, 32);
      const float mu = s * (1.0f / 32.0f);
      const float d0 = x0 - mu, d1 = x1 - mu;
      float q = d0 * d0 + d1 * d1;
#pragma unroll
      for (int o = 1; o < 16; o <<= 1) q += __shfl_xor(q, o, 32);
      const float inv = rsqrtf(q * (1.0f / 32.0f) + 1e-5f);
      const float y0 = fmaxf(gl0 * d0 * inv + bl0, 0.0f);
      const float y1 = fmaxf(gl1 * d1 * inv + bl1, 0.0f);
      if (row < N) {
        Out[(size_t)row * 32 + lr]      = y0;
        Out[(size_t)row * 32 + 16 + lr] = y1;
      }
    }
  } else if (epi == 2) {
    const float b0 = bv[lr];
    const float b1 = (lr < 4) ? bv[16 + lr] : 0.0f;
#pragma unroll
    for (int v = 0; v < 8; ++v) {
      const int row = m0 + half * 8 + v;
      if (row < N) {
        Out[(size_t)row * 20 + lr] = acc0[v] + b0;
        if (lr < 4) Out[(size_t)row * 20 + 16 + lr] = acc1[v] + b1;
      }
    }
  } else {
#pragma unroll
    for (int v = 0; v < 8; ++v) {
      const int row = m0 + half * 8 + v;
      if (row < N) {
        Out[(size_t)row * 32 + lr]      = acc0[v];
        Out[(size_t)row * 32 + 16 + lr] = acc1[v];
      }
    }
  }
}

// ---------------------------------------------------------------------------
// Edge-phase kernels (segment softmax over dst with atomics)
// ---------------------------------------------------------------------------
__device__ __forceinline__ int fOrd(float f) {
  int i = __float_as_int(f);
  return i < 0 ? (i ^ 0x7fffffff) : i;
}

__global__ void gather_rows(const float* __restrict__ X, const int* __restrict__ idx,
                            float* __restrict__ Out, int N) {
  int t = blockIdx.x * blockDim.x + threadIdx.x;
  if (t >= N * 8) return;
  int r = t >> 3, c = t & 7;
  ((float4*)Out)[t] = ((const float4*)(X + (size_t)idx[r] * 32))[c];
}

__global__ void attn_init(int* __restrict__ m, float* __restrict__ den,
                          float* __restrict__ F, const float* __restrict__ initF, int N) {
  int t = blockIdx.x * blockDim.x + threadIdx.x;
  if (t < N) { m[t] = 0x807fffff /* fOrd(-inf) */; den[t] = 0.0f; }
  if (t < N * 32) F[t] = initF ? initF[t] : 0.0f;
}

__global__ void edge_score(const float* __restrict__ Q, const float* __restrict__ K,
                           const int* __restrict__ src, const int* __restrict__ dst,
                           float* __restrict__ s, int* __restrict__ mbits, int E) {
  int e = blockIdx.x * blockDim.x + threadIdx.x;
  if (e >= E) return;
  const float4* q = (const float4*)(Q + (size_t)dst[e] * 32);
  const float4* k = (const float4*)(K + (size_t)src[e] * 32);
  float acc = 0.0f;
#pragma unroll
  for (int i = 0; i < 8; ++i) {
    float4 a = q[i], b = k[i];
    acc += a.x * b.x + a.y * b.y + a.z * b.z + a.w * b.w;
  }
  acc *= 0.17677669529663687f;  // 1/sqrt(32)
  s[e] = acc;
  atomicMax(mbits + dst[e], fOrd(acc));
}

__global__ void edge_exp(float* __restrict__ s, const int* __restrict__ dst,
                         const int* __restrict__ mbits, float* __restrict__ den, int E) {
  int e = blockIdx.x * blockDim.x + threadIdx.x;
  if (e >= E) return;
  int mb = mbits[dst[e]];
  float m = __int_as_float(mb < 0 ? (mb ^ 0x7fffffff) : mb);
  float ex = __expf(s[e] - m);
  s[e] = ex;
  atomicAdd(den + dst[e], ex);
}

__global__ void edge_accum(const float* __restrict__ s, const float* __restrict__ den,
                           const float* __restrict__ V, const int* __restrict__ src,
                           const int* __restrict__ dst, float* __restrict__ Out, int E) {
  int t = blockIdx.x * blockDim.x + threadIdx.x;
  int e = t >> 3;
  if (e >= E) return;
  int c = (t & 7) * 4;
  int d = dst[e];
  float coef = s[e] / (den[d] + 1e-9f);
  float4 v = *(const float4*)(V + (size_t)src[e] * 32 + c);
  float* o = Out + (size_t)d * 32 + c;
  atomicAdd(o + 0, coef * v.x);
  atomicAdd(o + 1, coef * v.y);
  atomicAdd(o + 2, coef * v.z);
  atomicAdd(o + 3, coef * v.w);
}

// ---------------------------------------------------------------------------
// Host orchestration
// ---------------------------------------------------------------------------
extern "C" void kernel_launch(void* const* d_in, const int* in_sizes, int n_in,
                              void* d_out, int out_size, void* d_ws, size_t ws_size,
                              hipStream_t stream) {
  const int NSs[7]  = {3125, 6250, 12500, 25000, 50000, 100000, 200000};  // mid,l5..l0
  const int CINs[7] = {224, 192, 160, 128, 96, 64, 32};

  const float* x_mid = (const float*)d_in[0];
  const float* yv[6]; for (int i = 0; i < 6; ++i) yv[i] = (const float*)d_in[1 + i];
  const int* edge[7]; for (int i = 0; i < 7; ++i) edge[i] = (const int*)d_in[7 + i];
  const int* trace[6]; for (int i = 0; i < 6; ++i) trace[i] = (const int*)d_in[14 + i];
  const int* p2v[7]; for (int i = 0; i < 7; ++i) p2v[i] = (const int*)d_in[20 + i];

  auto P = [&](int i) { return (const float*)d_in[i]; };

  // Detect pytree flattening of 'params' from leaf sizes.
  const int pb = 27;
  const bool midFirst = (in_sizes[pb] == 224 * 32);          // insertion-order top dict
  const int lvlBase   = midFirst ? pb + 8 : pb;
  const bool innerSorted = (in_sizes[lvlBase] != 192 * 32);  // sorted -> Wck(1024) first
  const int midBase   = midFirst ? pb : pb + 66;

  struct Lvl { const float *Wck,*Wcq,*Wcv,*Wd,*Wi,*Wk,*Wo,*Wq,*Wv,*bb,*gg; } L[6];
  for (int i = 0; i < 6; ++i) {
    int b = lvlBase + i * 11;
    if (innerSorted) {  // Wck,Wcq,Wcv,Wd,Wi,Wk,Wo,Wq,Wv,b,g
      L[i].Wck = P(b+0); L[i].Wcq = P(b+1); L[i].Wcv = P(b+2); L[i].Wd = P(b+3);
      L[i].Wi  = P(b+4); L[i].Wk  = P(b+5); L[i].Wo  = P(b+6); L[i].Wq = P(b+7);
      L[i].Wv  = P(b+8); L[i].bb  = P(b+9); L[i].gg  = P(b+10);
    } else {            // Wd,Wcq,Wck,Wcv,Wi,Wq,Wk,Wv,Wo,g,b
      L[i].Wd  = P(b+0); L[i].Wcq = P(b+1); L[i].Wck = P(b+2); L[i].Wcv = P(b+3);
      L[i].Wi  = P(b+4); L[i].Wq  = P(b+5); L[i].Wk  = P(b+6); L[i].Wv  = P(b+7);
      L[i].Wo  = P(b+8); L[i].gg  = P(b+9); L[i].bb  = P(b+10);
    }
  }
  struct { const float *Wd,*Wi,*Wk,*Wo,*Wq,*Wv,*bb,*gg; } M;
  if (innerSorted) {  // Wd,Wi,Wk,Wo,Wq,Wv,b,g
    M.Wd = P(midBase+0); M.Wi = P(midBase+1); M.Wk = P(midBase+2); M.Wo = P(midBase+3);
    M.Wq = P(midBase+4); M.Wv = P(midBase+5); M.bb = P(midBase+6); M.gg = P(midBase+7);
  } else {            // Wd,Wi,Wq,Wk,Wv,Wo,g,b
    M.Wd = P(midBase+0); M.Wi = P(midBase+1); M.Wq = P(midBase+2); M.Wk = P(midBase+3);
    M.Wv = P(midBase+4); M.Wo = P(midBase+5); M.gg = P(midBase+6); M.bb = P(midBase+7);
  }
  const float* Wout = P(n_in - 2);
  const float* bout = P(n_in - 1);

  // Workspace layout (floats)
  const size_t NMAX = 200000, EMAX = 1600000, FEAT = NMAX * 32;
  float* ws  = (float*)d_ws;
  float* GO  = ws + 0 * FEAT;
  float* GEO = ws + 1 * FEAT;
  float* Z   = ws + 2 * FEAT;
  float* QB  = ws + 3 * FEAT;
  float* KB  = ws + 4 * FEAT;
  float* VB  = ws + 5 * FEAT;
  float* FB  = ws + 6 * FEAT;
  float* GIB = ws + 7 * FEAT;
  float* S   = ws + 8 * FEAT;
  int*   MB  = (int*)(ws + 8 * FEAT + EMAX);
  float* DEN = ws + 8 * FEAT + EMAX + NMAX;

  auto gemm = [&](const float* X0, const int* G0, int cin0,
                  const float* X1, const float* X2, const float* W, int N,
                  float* Out, int epi, const float* gv, const float* bv) {
    wmma_gemm<<<dim3((N + 63) / 64), dim3(128), 0, stream>>>(
        X0, G0, cin0, X1, X1 ? 32 : 0, X2, X2 ? 32 : 0, W, N, Out, epi, gv, bv);
  };
  auto attn = [&](const float* Qp, const float* Kp, const float* Vp,
                  const int* ei, int E, int n, float* F, const float* initF) {
    attn_init<<<dim3((n * 32 + 255) / 256), dim3(256), 0, stream>>>(MB, DEN, F, initF, n);
    edge_score<<<dim3((E + 255) / 256), dim3(256), 0, stream>>>(Qp, Kp, ei, ei + E, S, MB, E);
    edge_exp<<<dim3((E + 255) / 256), dim3(256), 0, stream>>>(S, ei + E, MB, DEN, E);
    edge_accum<<<dim3((E * 8 + 255) / 256), dim3(256), 0, stream>>>(S, DEN, Vp, ei, ei + E, F, E);
  };

  // ----- mid level -----
  {
    const int n = NSs[0], E = 8 * n;
    gemm(x_mid, p2v[0], 224, nullptr, nullptr, M.Wd, n, Z, 0, nullptr, nullptr);
    gemm(Z, nullptr, 32, nullptr, nullptr, M.Wi, n, GIB, 0, nullptr, nullptr);
    gemm(GIB, nullptr, 32, nullptr, nullptr, M.Wq, n, QB, 0, nullptr, nullptr);
    gemm(GIB, nullptr, 32, nullptr, nullptr, M.Wk, n, KB, 0, nullptr, nullptr);
    gemm(GIB, nullptr, 32, nullptr, nullptr, M.Wv, n, VB, 0, nullptr, nullptr);
    attn(QB, KB, VB, edge[0], E, n, FB, GIB);               // residual folded into init
    gemm(FB, nullptr, 32, nullptr, nullptr, M.Wo, n, GO, 1, M.gg, M.bb);
  }

  // ----- decoder levels 5..0 -----
  for (int i = 0; i < 6; ++i) {
    const int n = NSs[1 + i], cin = CINs[1 + i], E = 8 * n;
    const Lvl& Lp = L[i];
    gather_rows<<<dim3((n * 8 + 255) / 256), dim3(256), 0, stream>>>(GO, trace[i], GEO, n);
    gemm(yv[i], p2v[1 + i], cin, nullptr, nullptr, Lp.Wd, n, Z, 0, nullptr, nullptr);
    // inter-domain attention: q from geo, k/v from z
    gemm(GEO, nullptr, 32, nullptr, nullptr, Lp.Wcq, n, QB, 0, nullptr, nullptr);
    gemm(Z,   nullptr, 32, nullptr, nullptr, Lp.Wck, n, KB, 0, nullptr, nullptr);
    gemm(Z,   nullptr, 32, nullptr, nullptr, Lp.Wcv, n, VB, 0, nullptr, nullptr);
    attn(QB, KB, VB, edge[1 + i], E, n, FB, nullptr);       // FB = at
    // gi = concat(geo, at, z) @ Wi
    gemm(GEO, nullptr, 32, FB, Z, Lp.Wi, n, GIB, 0, nullptr, nullptr);
    // intra attention + residual
    gemm(GIB, nullptr, 32, nullptr, nullptr, Lp.Wq, n, QB, 0, nullptr, nullptr);
    gemm(GIB, nullptr, 32, nullptr, nullptr, Lp.Wk, n, KB, 0, nullptr, nullptr);
    gemm(GIB, nullptr, 32, nullptr, nullptr, Lp.Wv, n, VB, 0, nullptr, nullptr);
    attn(QB, KB, VB, edge[1 + i], E, n, FB, GIB);           // init FB = gi (residual)
    gemm(FB, nullptr, 32, nullptr, nullptr, Lp.Wo, n, GO, 1, Lp.gg, Lp.bb);
  }

  // ----- output head: [200000,32] @ [32,20] + b -----
  gemm(GO, nullptr, 32, nullptr, nullptr, Wout, NSs[6], (float*)d_out, 2, nullptr, bout);
}
